// KerSr_17197049053612
// MI455X (gfx1250) — matrix operand order repaired and checked by reading, not compiled, not run on hardware
//
#include <hip/hip_runtime.h>
#include <stdint.h>

// ============================================================================
// SR network on MI455X (gfx1250): implicit-GEMM 3x3 convs via
// V_WMMA_F32_16X16X32_F16, activations NHWC f16, fp32 accumulate.
//  - templated <CinP,CoutP>: fully unrolled K/N loops, static accumulators
//  - input slab (3 rows x 130 px x CinP) staged to LDS per block with
//    GLOBAL_LOAD_ASYNC_TO_LDS_B128 (ASYNCcnt), halo zero-filled -> branch-free
//    inner loop of ds_load_b128 + v_wmma, and 3x (not 9x) global A traffic
//  - packed weights also async-staged to LDS
// ============================================================================

typedef __attribute__((ext_vector_type(16))) _Float16 v16h;
typedef __attribute__((ext_vector_type(8)))  float    v8f;
typedef __attribute__((ext_vector_type(4)))  uint32_t v4u;

union AFrag { v16h h; v4u u4[2]; };

__device__ __forceinline__ void async_g2l_b128(uint32_t lds_off, const void* gaddr) {
  asm volatile("global_load_async_to_lds_b128 %0, %1, off"
               :: "v"(lds_off), "v"(gaddr) : "memory");
}
__device__ __forceinline__ void wait_async0() {
  asm volatile("s_wait_asynccnt 0x0" ::: "memory");
}

// ---------------------------------------------------------------------------
// Pack OIHW fp32 weights into WMMA B-fragment-ready f16 layout.
// Fragment index = ((r*3+s)*nCC + cc)*nNT + nt, each fragment = 512 f16:
//   element (lane L, slot idx): n = L%16, half = L/16, K = 16*half + idx
//   value = W[co = nt*16+n][ci = cc*32+K][r][s]  (zero outside real Cin/Cout)
// ---------------------------------------------------------------------------
__global__ void pack_weights(const float* __restrict__ w, _Float16* __restrict__ dst,
                             int CinP, int CoutP, int CinR, int CoutR) {
  int nCC = CinP >> 5, nNT = CoutP >> 4;
  int total = 9 * nCC * nNT * 512;
  int e = blockIdx.x * blockDim.x + threadIdx.x;
  if (e >= total) return;
  int frag = e >> 9;
  int rem  = e & 511;
  int lane = rem >> 4;
  int idx  = rem & 15;
  int half = lane >> 4;
  int n    = lane & 15;
  int K    = half * 16 + idx;
  int nt   = frag % nNT;
  int cc   = (frag / nNT) % nCC;
  int rs   = frag / (nNT * nCC);
  int co   = nt * 16 + n;
  int ci   = cc * 32 + K;
  float v = 0.f;
  if (co < CoutR && ci < CinR)
    v = w[((long)co * CinR + ci) * 9 + rs];
  dst[e] = (_Float16)v;
}

// ---------------------------------------------------------------------------
// Direct 3x3 conv for Cin==1 (e1, k1): fp32 math, ReLU, NHWC f16 out.
// ---------------------------------------------------------------------------
template<int CO, int COP>
__global__ void conv_c1_direct(const float* __restrict__ x, const float* __restrict__ w,
                               _Float16* __restrict__ out, int Nn, int H, int W) {
  long total = (long)Nn * H * W;
  long p = (long)blockIdx.x * blockDim.x + threadIdx.x;
  if (p >= total) return;
  int xx = (int)(p % W);
  int yy = (int)((p / W) % H);
  int n  = (int)(p / ((long)W * H));
  float acc[CO];
#pragma unroll
  for (int c = 0; c < CO; ++c) acc[c] = 0.f;
  for (int r = 0; r < 3; ++r) {
    int y2 = yy + r - 1;
    if ((unsigned)y2 >= (unsigned)H) continue;
    for (int s = 0; s < 3; ++s) {
      int x2 = xx + s - 1;
      if ((unsigned)x2 >= (unsigned)W) continue;
      float v = x[((long)n * H + y2) * W + x2];
#pragma unroll
      for (int c = 0; c < CO; ++c) acc[c] += v * w[c * 9 + r * 3 + s];
    }
  }
  _Float16* o = out + p * COP;
#pragma unroll
  for (int c = 0; c < CO; ++c) o[c] = (_Float16)(acc[c] > 0.f ? acc[c] : 0.f);
#pragma unroll
  for (int c = CO; c < COP; ++c) o[c] = (_Float16)0.f;
}

// ---------------------------------------------------------------------------
// Implicit-GEMM 3x3 SAME conv with V_WMMA_F32_16X16X32_F16.
// Block = 256 threads = 8 waves; one 128-pixel row segment per block, each
// wave owns 16 consecutive pixels x full CoutP. Input slab and packed weights
// async-staged to LDS; inner loop is branch-free ds_load + wmma.
// ---------------------------------------------------------------------------
template<int CinP, int CoutP>
__global__ __launch_bounds__(256)
void conv3x3_wmma(const _Float16* __restrict__ in, _Float16* __restrict__ out,
                  const _Float16* __restrict__ wB,
                  int Nn, int H, int W) {
  constexpr int nCC = CinP / 32;               // 32-wide K chunks
  constexpr int nNT = CoutP / 16;              // 16-wide N tiles (<= 4)
  constexpr int wFragBytes = 9 * nCC * nNT * 1024;
  constexpr int slabPix    = 130;              // 128 + 2 halo
  constexpr int pixBytes   = CinP * 2;
  constexpr int pixChunks  = pixBytes / 16;    // 16B chunks per pixel
  __shared__ __align__(16) char smemW[wFragBytes];
  __shared__ __align__(16) char smemIn[3 * slabPix * pixBytes];

  const int segs = W >> 7;                     // 128-pixel segments per row
  int blk = blockIdx.x;
  int n0 = blk / (H * segs);
  int r2 = blk % (H * segs);
  int y  = r2 / segs;
  int x0blk = (r2 % segs) * 128;

  // ---- async-stage packed weights (all in-bounds) ----
  for (int i = threadIdx.x; i < wFragBytes / 16; i += 256) {
    uint32_t loff = (uint32_t)(uintptr_t)(smemW + i * 16);
    async_g2l_b128(loff, (const char*)wB + i * 16);
  }

  // ---- async-stage input slab rows y-1..y+1, cols x0blk-1..x0blk+128 ----
  // OOB cells zero-filled with plain LDS stores (DScnt, covered by barrier).
  {
    const int total = 3 * slabPix * pixChunks;
    for (int t = threadIdx.x; t < total; t += 256) {
      int ch = t % pixChunks;
      int p  = (t / pixChunks) % slabPix;
      int r  = t / (pixChunks * slabPix);
      int gy = y - 1 + r;
      int gx = x0blk - 1 + p;
      char* l = smemIn + (size_t)(r * slabPix + p) * pixBytes + ch * 16;
      if ((unsigned)gy < (unsigned)H && (unsigned)gx < (unsigned)W) {
        const char* g = (const char*)in
            + ((((long)n0 * H + gy) * W + gx) * CinP) * 2 + ch * 16;
        async_g2l_b128((uint32_t)(uintptr_t)l, g);
      } else {
        *(v4u*)l = (v4u){0u, 0u, 0u, 0u};
      }
    }
  }
  wait_async0();
  __syncthreads();

  const int wave = threadIdx.x >> 5;
  const int lane = threadIdx.x & 31;
  const int half = lane >> 4;
  const int m    = lane & 15;
  const int x0   = x0blk + wave * 16;

  v8f acc[nNT];
#pragma unroll
  for (int t = 0; t < nNT; ++t) acc[t] = (v8f){0.f,0.f,0.f,0.f,0.f,0.f,0.f,0.f};

#pragma unroll
  for (int r = 0; r < 3; ++r) {
#pragma unroll
    for (int s = 0; s < 3; ++s) {
      // lane's pixel in slab: col0 of slab = x0blk-1  ->  index wave*16+m+s
      const int p = wave * 16 + m + s;
      const char* pixBase = smemIn + (size_t)(r * slabPix + p) * pixBytes + half * 32;
#pragma unroll
      for (int cc = 0; cc < nCC; ++cc) {
        AFrag a;
        // A layout: lane<16 -> K 0..15 ; lane>=16 -> K 16..31
        a.u4[0] = *(const v4u*)(pixBase + cc * 64);
        a.u4[1] = *(const v4u*)(pixBase + cc * 64 + 16);
        const int fbase = ((r * 3 + s) * nCC + cc) * nNT;
#pragma unroll
        for (int nt = 0; nt < nNT; ++nt) {
          const v16h* bp = (const v16h*)(smemW + (size_t)(fbase + nt) * 1024 + lane * 32);
          acc[nt] = __builtin_amdgcn_wmma_f32_16x16x32_f16(
              false, a.h, false, *bp, (short)0, acc[nt], false, false);
        }
      }
    }
  }

  // C/D layout: VGPR vr, lanes 0-15 -> M=vr, lanes 16-31 -> M=vr+8; N = lane%16
  long rowBase = ((long)n0 * H + y) * W;
#pragma unroll
  for (int nt = 0; nt < nNT; ++nt) {
    int co = nt * 16 + m;
#pragma unroll
    for (int vr = 0; vr < 8; ++vr) {
      int mm = vr + half * 8;
      float v = acc[nt][vr];
      v = v > 0.f ? v : 0.f;
      out[(rowBase + x0 + mm) * CoutP + co] = (_Float16)v;
    }
  }
}

// ---------------------------------------------------------------------------
// PixelShuffle x4 + ReLU: a5 NHWC(16ch f16, 256^2) -> h fp32 (1024^2).
// out[n, 4h+i, 4w+j] = relu(a5[n,h,w, c=i*4+j])
// ---------------------------------------------------------------------------
__global__ void pixshuf_relu(const _Float16* __restrict__ a5, float* __restrict__ h,
                             int Nn, int H, int W) {
  int HO = H * 4, WO = W * 4;
  long total = (long)Nn * HO * WO;
  long p = (long)blockIdx.x * blockDim.x + threadIdx.x;
  if (p >= total) return;
  int X = (int)(p % WO);
  int Y = (int)((p / WO) % HO);
  int n = (int)(p / ((long)WO * HO));
  int hh = Y >> 2, i = Y & 3, ww = X >> 2, j = X & 3;
  float v = (float)a5[(((long)n * H + hh) * W + ww) * 16 + (i * 4 + j)];
  h[p] = v > 0.f ? v : 0.f;
}

// ---------------------------------------------------------------------------
// Per-pixel 5x5 dynamic filter + residual: out = h + sum_j h_pad[...]*z[j]
// kernel channel j -> (kw=j/5, kh=j%5); tap = h[y+kh-2, x+kw-2]
// ---------------------------------------------------------------------------
__global__ void pixconv_add(const float* __restrict__ h, const _Float16* __restrict__ z,
                            float* __restrict__ out, int Nn, int H, int W) {
  long total = (long)Nn * H * W;
  long p = (long)blockIdx.x * blockDim.x + threadIdx.x;
  if (p >= total) return;
  int xx = (int)(p % W);
  int yy = (int)((p / W) % H);
  int n  = (int)(p / ((long)W * H));
  const float* hn = h + (long)n * H * W;
  const _Float16* zp = z + p * 32;
  float acc = 0.f;
#pragma unroll
  for (int j = 0; j < 25; ++j) {
    int kw = j / 5, kh = j % 5;
    int y2 = yy + kh - 2, x2 = xx + kw - 2;
    float hv = ((unsigned)y2 < (unsigned)H && (unsigned)x2 < (unsigned)W)
                   ? hn[(long)y2 * W + x2] : 0.f;
    acc += hv * (float)zp[j];
  }
  out[p] = h[p] + acc;
}

// ---------------------------------------------------------------------------
// Host orchestration
// ---------------------------------------------------------------------------
static inline size_t wbytes(int CinP, int CoutP) {
  return (size_t)9 * (CinP / 32) * (CoutP / 16) * 512 * 2;
}

extern "C" void kernel_launch(void* const* d_in, const int* in_sizes, int n_in,
                              void* d_out, int out_size, void* d_ws, size_t ws_size,
                              hipStream_t stream) {
  (void)in_sizes; (void)n_in; (void)out_size; (void)ws_size;
  const float* x    = (const float*)d_in[0];
  const float* w_e1 = (const float*)d_in[1];
  const float* w_e2 = (const float*)d_in[2];
  const float* w_e3 = (const float*)d_in[3];
  const float* w_d1 = (const float*)d_in[4];
  const float* w_d2 = (const float*)d_in[5];
  const float* w_k1 = (const float*)d_in[6];
  const float* w_k2 = (const float*)d_in[7];
  const float* w_k3 = (const float*)d_in[8];
  const float* w_k4 = (const float*)d_in[9];
  float* out = (float*)d_out;
  char* ws = (char*)d_ws;

  const int N = 2, H1 = 256, W1 = 256, H2 = 1024, W2 = 1024;
  const long P1 = (long)N * H1 * W1;   // 131072
  const long P2 = (long)N * H2 * W2;   // 2097152

  // ---- workspace layout (byte offsets, 256-aligned) ----
  size_t off = 0;
  auto alloc = [&](size_t bytes) { size_t o = off; off = (off + bytes + 255) & ~(size_t)255; return o; };
  size_t oWe2 = alloc(wbytes(32, 32));
  size_t oWe3 = alloc(wbytes(32, 64));
  size_t oWd1 = alloc(wbytes(64, 32));
  size_t oWd2 = alloc(wbytes(32, 16));
  size_t oWk2 = alloc(wbytes(32, 64));
  size_t oWk3 = alloc(wbytes(64, 32));
  size_t oWk4 = alloc(wbytes(32, 32));
  size_t oA1 = alloc((size_t)P1 * 32 * 2);
  size_t oA2 = alloc((size_t)P1 * 32 * 2);
  size_t oA3 = alloc((size_t)P1 * 64 * 2);
  size_t oA4 = alloc((size_t)P1 * 32 * 2);
  size_t oA5 = alloc((size_t)P1 * 16 * 2);
  size_t oH  = alloc((size_t)P2 * 4);
  size_t oB1 = alloc((size_t)P2 * 32 * 2);   // k1 out / k3 out (reused)
  size_t oB2 = alloc((size_t)P2 * 64 * 2);   // k2 out / k4 out (reused)

  _Float16* wBe2 = (_Float16*)(ws + oWe2);
  _Float16* wBe3 = (_Float16*)(ws + oWe3);
  _Float16* wBd1 = (_Float16*)(ws + oWd1);
  _Float16* wBd2 = (_Float16*)(ws + oWd2);
  _Float16* wBk2 = (_Float16*)(ws + oWk2);
  _Float16* wBk3 = (_Float16*)(ws + oWk3);
  _Float16* wBk4 = (_Float16*)(ws + oWk4);
  _Float16* a1 = (_Float16*)(ws + oA1);
  _Float16* a2 = (_Float16*)(ws + oA2);
  _Float16* a3 = (_Float16*)(ws + oA3);
  _Float16* a4 = (_Float16*)(ws + oA4);
  _Float16* a5 = (_Float16*)(ws + oA5);
  float*    hB = (float*)(ws + oH);
  _Float16* b1 = (_Float16*)(ws + oB1);
  _Float16* b2 = (_Float16*)(ws + oB2);
  _Float16* b3 = b1;                         // reuse: b1 dead after k2
  _Float16* zB = b2;                         // reuse: b2 dead after k3

  // ---- weight packing ----
  auto packGrid = [](int CinP, int CoutP) {
    int total = 9 * (CinP / 32) * (CoutP / 16) * 512;
    return (total + 255) / 256;
  };
  pack_weights<<<packGrid(32, 32), 256, 0, stream>>>(w_e2, wBe2, 32, 32, 16, 32);
  pack_weights<<<packGrid(32, 64), 256, 0, stream>>>(w_e3, wBe3, 32, 64, 32, 64);
  pack_weights<<<packGrid(64, 32), 256, 0, stream>>>(w_d1, wBd1, 64, 32, 64, 32);
  pack_weights<<<packGrid(32, 16), 256, 0, stream>>>(w_d2, wBd2, 32, 16, 32, 16);
  pack_weights<<<packGrid(32, 64), 256, 0, stream>>>(w_k2, wBk2, 32, 64, 32, 64);
  pack_weights<<<packGrid(64, 32), 256, 0, stream>>>(w_k3, wBk3, 64, 32, 64, 32);
  pack_weights<<<packGrid(32, 32), 256, 0, stream>>>(w_k4, wBk4, 32, 32, 32, 25);

  // ---- stage A: 256^2 ----
  conv_c1_direct<16, 32><<<(int)((P1 + 255) / 256), 256, 0, stream>>>(x, w_e1, a1, N, H1, W1);

  auto convGrid = [](int Nn, int H, int W) { return Nn * H * (W >> 7); };
  conv3x3_wmma<32, 32><<<convGrid(N, H1, W1), 256, 0, stream>>>(a1, a2, wBe2, N, H1, W1);
  conv3x3_wmma<32, 64><<<convGrid(N, H1, W1), 256, 0, stream>>>(a2, a3, wBe3, N, H1, W1);
  conv3x3_wmma<64, 32><<<convGrid(N, H1, W1), 256, 0, stream>>>(a3, a4, wBd1, N, H1, W1);
  conv3x3_wmma<32, 16><<<convGrid(N, H1, W1), 256, 0, stream>>>(a4, a5, wBd2, N, H1, W1);

  // ---- pixel shuffle x4 + relu -> h (fp32, 1024^2) ----
  pixshuf_relu<<<(int)((P2 + 255) / 256), 256, 0, stream>>>(a5, hB, N, H1, W1);

  // ---- stage B: 1024^2 kernel-prediction branch ----
  conv_c1_direct<32, 32><<<(int)((P2 + 255) / 256), 256, 0, stream>>>(hB, w_k1, b1, N, H2, W2);
  conv3x3_wmma<32, 64><<<convGrid(N, H2, W2), 256, 0, stream>>>(b1, b2, wBk2, N, H2, W2);
  conv3x3_wmma<64, 32><<<convGrid(N, H2, W2), 256, 0, stream>>>(b2, b3, wBk3, N, H2, W2);
  conv3x3_wmma<32, 32><<<convGrid(N, H2, W2), 256, 0, stream>>>(b3, zB, wBk4, N, H2, W2);

  // ---- dynamic 5x5 filter + residual ----
  pixconv_add<<<(int)((P2 + 255) / 256), 256, 0, stream>>>(hB, zB, out, N, H2, W2);
}
